// MultiHeadAttention_89893665506084
// MI455X (gfx1250) — matrix operand order repaired and checked by reading, not compiled, not run on hardware
//
#include <hip/hip_runtime.h>

typedef __attribute__((ext_vector_type(16))) __bf16    v16bf;
typedef __attribute__((ext_vector_type(8)))  float     v8f;
typedef __attribute__((ext_vector_type(4)))  unsigned  u32x4;
typedef __attribute__((ext_vector_type(4)))  float     f32x4;

#define BB 2
#define SS 2048
#define DD 1024
#define HH 16
#define HD 64
#define NT (SS / 16)  /* 128 16-row seq tiles */
#define NT2 (SS / 32) /* 64 32-row seq tiles  */

union BF16x16 {
  v16bf v;
  unsigned short u[16];
  unsigned w[8];
  u32x4 q[2];
};

// f32 -> bf16, round-half-up (1 VALU; pairs well with store_d16_hi)
__device__ __forceinline__ unsigned short f2bf(float f) {
  union { float f; unsigned u; } c;
  c.f = f;
  return (unsigned short)((c.u + 0x8000u) >> 16);
}

// Pack two f32 into two bf16 (lo=a, hi=b), round-half-up.
__device__ __forceinline__ unsigned pack2bf(float a, float b) {
  union { float f; unsigned u; } ca, cb;
  ca.f = a;
  cb.f = b;
#if __has_builtin(__builtin_amdgcn_perm)
  // v_perm_b32: result = {hi16(cb+rnd), hi16(ca+rnd)} in one instruction
  return __builtin_amdgcn_perm(cb.u + 0x8000u, ca.u + 0x8000u, 0x07060302u);
#else
  return ((ca.u + 0x8000u) >> 16) | ((cb.u + 0x8000u) & 0xffff0000u);
#endif
}

// Raw v_exp_f32 (args are range-safe here; avoids libm fixup code)
__device__ __forceinline__ float fast_exp2(float x) {
#if __has_builtin(__builtin_amdgcn_exp2f)
  return __builtin_amdgcn_exp2f(x);
#else
  return exp2f(x);
#endif
}

// Raw v_rcp_f32 (1 ulp; fine for softmax normalization)
__device__ __forceinline__ float fast_rcp(float x) {
#if __has_builtin(__builtin_amdgcn_rcpf)
  return __builtin_amdgcn_rcpf(x);
#else
  return 1.0f / x;
#endif
}

__device__ __forceinline__ v8f wmma_bf16(v16bf a, v16bf b, v8f c) {
  // D = A(16x32 bf16) x B(32x16 bf16) + C(16x16 f32)
  return __builtin_amdgcn_wmma_f32_16x16x32_bf16(
      false, a, false, b, (short)0, c, false, false);
}

// A-fragment (16x32 bf16): lane m = lane&15; K contiguous 16 at hi*16.
__device__ __forceinline__ void load_a_frag(BF16x16& a,
                                            const unsigned short* row, int hi) {
  const u32x4* p = reinterpret_cast<const u32x4*>(row);
  a.q[0] = p[2 * hi];
  a.q[1] = p[2 * hi + 1];
}

// B-fragment (32x16 bf16): lane's 16 values are CONTIGUOUS starting at run16.
__device__ __forceinline__ void load_b_frag(BF16x16& b,
                                            const unsigned short* run16) {
  const u32x4* p = reinterpret_cast<const u32x4*>(run16);
  b.q[0] = p[0];
  b.q[1] = p[1];
}

constexpr float SCL = 0.125f * 1.44269504088896f;  // 1/sqrt(64) * log2(e)
constexpr float OFS = -32.0f;  // static softmax offset (scores bounded)

// ---------------------------------------------------------------------------
// Kernel 0a: bf16 convert + transpose inner 64x64:  dst[h][o][i]=src[h][i][o]
// ---------------------------------------------------------------------------
__global__ void __launch_bounds__(256) cvt_t64(
    const float* __restrict__ src, unsigned short* __restrict__ dst) {
  const int i = blockIdx.x * blockDim.x + threadIdx.x;  // H*64*64
  const int h = i >> 12, rem = i & 4095;
  const int din = rem >> 6, o = rem & 63;
  dst[h * 4096 + o * 64 + din] = f2bf(src[i]);
}

// ---------------------------------------------------------------------------
// Kernel 0b: bf16 convert + transpose DxD:  dst[n][k] = src[k][n]
// ---------------------------------------------------------------------------
__global__ void __launch_bounds__(256) cvt_tD(
    const float* __restrict__ src, unsigned short* __restrict__ dst) {
  const int i = blockIdx.x * blockDim.x + threadIdx.x;  // D*D
  const int k = i >> 10, n = i & 1023;
  dst[n * DD + k] = f2bf(src[i]);
}

// ---------------------------------------------------------------------------
// Kernel 1: per-head QKV projection.  One wave -> 16 seq rows x 64 dims.
// Weights pre-transposed bf16 [H][out][in] -> contiguous B-fragments.
// Q,K row-major [B,H,S,64]; V transposed [B,H,64,S].
// ---------------------------------------------------------------------------
__device__ __forceinline__ void gemm16x64(const BF16x16 a[2],
                                          const unsigned short* __restrict__ Wt,
                                          int nlane, int hi, v8f c[4]) {
#pragma unroll
  for (int nt = 0; nt < 4; ++nt) {
    v8f acc = {};
#pragma unroll
    for (int kb = 0; kb < 2; ++kb) {
      BF16x16 bw;  // B[k][n] = Wt[n][k]
      load_b_frag(bw, Wt + (nt * 16 + nlane) * HD + kb * 32 + hi * 16);
      acc = wmma_bf16(a[kb].v, bw.v, acc);
    }
    c[nt] = acc;
  }
}

__global__ void __launch_bounds__(256) qkv_proj(
    const float* __restrict__ x,
    const unsigned short* __restrict__ Wqt,
    const unsigned short* __restrict__ Wkt,
    const unsigned short* __restrict__ Wvt,
    unsigned short* __restrict__ Qw,
    unsigned short* __restrict__ Kw,
    unsigned short* __restrict__ Vt) {
  const int lane = threadIdx.x & 31;
  const int wid  = blockIdx.x * (blockDim.x >> 5) + (threadIdx.x >> 5);
  if (wid >= BB * HH * NT) return;  // exact grid
  const int b  = wid / (HH * NT);
  const int h  = (wid / NT) % HH;
  const int s0 = (wid % NT) * 16;
  const int nlane = lane & 15;
  const int hi    = lane >> 4;

  // x tile -> bf16 A-fragments (vector loads + packed converts)
  BF16x16 a[2];
  {
    const float* xrow = x + ((size_t)(b * SS + s0 + nlane)) * DD + h * HD;
#pragma unroll
    for (int kb = 0; kb < 2; ++kb) {
      const f32x4* px = reinterpret_cast<const f32x4*>(xrow + kb * 32);
      f32x4 c0 = px[4 * hi], c1 = px[4 * hi + 1];
      f32x4 c2 = px[4 * hi + 2], c3 = px[4 * hi + 3];
      a[kb].w[0] = pack2bf(c0[0], c0[1]);
      a[kb].w[1] = pack2bf(c0[2], c0[3]);
      a[kb].w[2] = pack2bf(c1[0], c1[1]);
      a[kb].w[3] = pack2bf(c1[2], c1[3]);
      a[kb].w[4] = pack2bf(c2[0], c2[1]);
      a[kb].w[5] = pack2bf(c2[2], c2[3]);
      a[kb].w[6] = pack2bf(c3[0], c3[1]);
      a[kb].w[7] = pack2bf(c3[2], c3[3]);
    }
  }

  const int hb = (b * HH + h);
  v8f c[4];

  gemm16x64(a, Wqt + h * HD * HD, nlane, hi, c);  // ---- Q ----
  {
    unsigned short* O = Qw + ((size_t)(hb * SS + s0)) * HD;
#pragma unroll
    for (int nt = 0; nt < 4; ++nt)
#pragma unroll
      for (int r = 0; r < 8; ++r)
        O[(r + 8 * hi) * HD + nt * 16 + nlane] = f2bf(c[nt][r]);
  }
  gemm16x64(a, Wkt + h * HD * HD, nlane, hi, c);  // ---- K ----
  {
    unsigned short* O = Kw + ((size_t)(hb * SS + s0)) * HD;
#pragma unroll
    for (int nt = 0; nt < 4; ++nt)
#pragma unroll
      for (int r = 0; r < 8; ++r)
        O[(r + 8 * hi) * HD + nt * 16 + nlane] = f2bf(c[nt][r]);
  }
  gemm16x64(a, Wvt + h * HD * HD, nlane, hi, c);  // ---- V (transposed) ----
  {
    unsigned short* O = Vt + (size_t)hb * HD * SS + s0;
#pragma unroll
    for (int nt = 0; nt < 4; ++nt)
#pragma unroll
      for (int r = 0; r < 8; ++r)
        O[(size_t)(nt * 16 + nlane) * SS + r + 8 * hi] = f2bf(c[nt][r]);
  }
}

// ---------------------------------------------------------------------------
// Kernel 2: causal flash attention, static-max softmax.
// One wave owns 32 queries (two 16-row tiles) of one (b,h): every K/V
// B-fragment is loaded once and used twice.  Row-sums (softmax denominator)
// accumulate through a 5th PV tile with B = ones.  Only the single diagonal
// step needs causal masks.
// ---------------------------------------------------------------------------
template <bool MASKED>
__device__ __forceinline__ void fa_step(
    int t0, int s0, int nlane, int hi,
    const unsigned short* __restrict__ Kh,   // K head base [S][64]
    const unsigned short* __restrict__ Vh,   // V head base [64][S]
    const BF16x16 aq[2][2], const BF16x16& bones,
    float* __restrict__ pl, v8f acc[2][5]) {
  // ---- scores: 2 key subtiles x 2 query tiles; K-frags loaded once ----
  v8f sc[2][2];
#pragma unroll
  for (int js = 0; js < 2; ++js) {
    const int t = t0 + js * 16 + nlane;
    const unsigned short* krow = Kh + (size_t)t * HD + hi * 16;
    BF16x16 bk0, bk1;
    load_b_frag(bk0, krow);
    load_b_frag(bk1, krow + 32);
#pragma unroll
    for (int qt = 0; qt < 2; ++qt) {
      v8f c = {};
      c = wmma_bf16(aq[qt][0].v, bk0.v, c);
      c = wmma_bf16(aq[qt][1].v, bk1.v, c);
      sc[qt][js] = c;
    }
  }
  // ---- static-max softmax numerators; stage to LDS f32 tiles ----
#pragma unroll
  for (int qt = 0; qt < 2; ++qt) {
    float* pt = pl + qt * 512;
#pragma unroll
    for (int r = 0; r < 8; ++r) {
      const int ml = r + 8 * hi;
      float e0 = fast_exp2(fmaf(sc[qt][0][r], SCL, OFS));
      float e1 = fast_exp2(fmaf(sc[qt][1][r], SCL, OFS));
      if (MASKED) {
        const int mr = s0 + qt * 16 + ml;  // global query row
        e0 = (t0 + nlane      <= mr) ? e0 : 0.0f;
        e1 = (t0 + 16 + nlane <= mr) ? e1 : 0.0f;
      }
      pt[ml * 32 + nlane]      = e0;
      pt[ml * 32 + 16 + nlane] = e1;
    }
  }
  // ---- rebuild P as bf16 A-fragments (ds_load_b64 + pack) ----
  BF16x16 ap[2];
#pragma unroll
  for (int qt = 0; qt < 2; ++qt) {
    const float* prow = pl + qt * 512 + nlane * 32;
#pragma unroll
    for (int j = 0; j < 8; ++j) {
      const int K = 2 * j + hi * 16;
      ap[qt].w[j] = pack2bf(prow[K], prow[K + 1]);
    }
  }
  // ---- acc += P x [V | ones]; V-frags loaded once, used twice ----
  const unsigned short* vrun = Vh + (size_t)nlane * SS + t0 + hi * 16;
#pragma unroll
  for (int dt = 0; dt < 4; ++dt) {
    BF16x16 bv;
    load_b_frag(bv, vrun + (size_t)(dt * 16) * SS);
    acc[0][dt] = wmma_bf16(ap[0].v, bv.v, acc[0][dt]);
    acc[1][dt] = wmma_bf16(ap[1].v, bv.v, acc[1][dt]);
  }
  acc[0][4] = wmma_bf16(ap[0].v, bones.v, acc[0][4]);
  acc[1][4] = wmma_bf16(ap[1].v, bones.v, acc[1][4]);
}

__global__ void __launch_bounds__(256) flash_attn(
    const unsigned short* __restrict__ Qw,   // [B,H,S,64]
    const unsigned short* __restrict__ Kw,   // [B,H,S,64]
    const unsigned short* __restrict__ Vt,   // [B,H,64,S]
    unsigned short* __restrict__ attn) {     // [B,S,D] channel = hd*H + h
  __shared__ float plds[8 * 1024];           // per wave: two 16x32 f32 P tiles
  const int lane  = threadIdx.x & 31;
  const int wslot = threadIdx.x >> 5;
  const int wid   = blockIdx.x * (blockDim.x >> 5) + wslot;
  if (wid >= BB * HH * NT2) return;
  const int b  = wid / (HH * NT2);
  const int h  = (wid / NT2) % HH;
  const int s0 = (wid % NT2) * 32;
  const int nlane = lane & 15;
  const int hi    = lane >> 4;

  const int hb = b * HH + h;
  const unsigned short* Kh = Kw + (size_t)hb * SS * HD;
  const unsigned short* Vh = Vt + (size_t)hb * HD * SS;

  BF16x16 aq[2][2];
#pragma unroll
  for (int qt = 0; qt < 2; ++qt) {
    const unsigned short* qrow =
        Qw + (size_t)hb * SS * HD + (size_t)(s0 + qt * 16 + nlane) * HD;
    load_a_frag(aq[qt][0], qrow, hi);
    load_a_frag(aq[qt][1], qrow + 32, hi);
  }

  BF16x16 bones;  // constant ones B-fragment -> row-sum accumulator
#pragma unroll
  for (int e = 0; e < 16; ++e) bones.u[e] = 0x3f80;

  v8f acc[2][5] = {};  // [qt][0..3]=PV tiles, [4]=row-sum (l)
  float* pl = plds + wslot * 1024;

  // Mask-free bulk: keys [0, s0)
  for (int t0 = 0; t0 < s0; t0 += 32) {
    __builtin_prefetch(Kh + (size_t)(t0 + 32) * HD, 0, 0);
    __builtin_prefetch(Vh + (size_t)nlane * SS + t0 + 32, 0, 0);
    fa_step<false>(t0, s0, nlane, hi, Kh, Vh, aq, bones, pl, acc);
  }
  // Single masked diagonal step: keys [s0, s0+32)
  fa_step<true>(s0, s0, nlane, hi, Kh, Vh, aq, bones, pl, acc);

  // ---- epilogue: normalize by l = acc[qt][4], store interleaved ----
#pragma unroll
  for (int qt = 0; qt < 2; ++qt)
#pragma unroll
    for (int r = 0; r < 8; ++r) {
      const float inv = fast_rcp(acc[qt][4][r]);
      unsigned short* orow =
          attn + ((size_t)(b * SS + s0 + qt * 16 + r + 8 * hi)) * DD;
#pragma unroll
      for (int dt = 0; dt < 4; ++dt)
        orow[(dt * 16 + nlane) * HH + h] = f2bf(acc[qt][dt][r] * inv);
    }
}

// ---------------------------------------------------------------------------
// Kernel 3: output projection y = attn @ Wo + bo.  One wave -> 16x64 tile.
// ---------------------------------------------------------------------------
__global__ void __launch_bounds__(256) out_proj(
    const unsigned short* __restrict__ ab,     // [B*S, D] bf16
    const unsigned short* __restrict__ WoT,    // [D out][D in] bf16
    const float* __restrict__ bo,
    float* __restrict__ out) {
  const int lane = threadIdx.x & 31;
  const int wid  = blockIdx.x * (blockDim.x >> 5) + (threadIdx.x >> 5);
  if (wid >= (BB * SS / 16) * (DD / 64)) return;
  const int rt = wid >> 4;
  const int ct = wid & 15;
  const int r0 = rt * 16, c0 = ct * 64;
  const int nlane = lane & 15, hi = lane >> 4;

  v8f c[4] = {};
  const unsigned short* arow = ab + (size_t)(r0 + nlane) * DD;
  for (int kb = 0; kb < 32; ++kb) {
    BF16x16 a;
    load_a_frag(a, arow + kb * 32, hi);
#pragma unroll
    for (int dt = 0; dt < 4; ++dt) {
      BF16x16 bw;  // B[k][n] = WoT[n][k]
      load_b_frag(bw, WoT + (size_t)(c0 + dt * 16 + nlane) * DD +
                          kb * 32 + hi * 16);
      c[dt] = wmma_bf16(a.v, bw.v, c[dt]);
    }
  }
#pragma unroll
  for (int dt = 0; dt < 4; ++dt) {
    const float bias = bo[c0 + dt * 16 + nlane];
#pragma unroll
    for (int r = 0; r < 8; ++r)
      out[(size_t)(r0 + r + 8 * hi) * DD + c0 + dt * 16 + nlane] =
          c[dt][r] + bias;
  }
}

// ---------------------------------------------------------------------------
// Launch: inputs = {x, Wq, Wk, Wv, Wo, bo} (all fp32)
// Workspace (bytes): Qw 8M | Kw 8M | Vt 8M | attn 8M | Wq/k/v^T 384K | Wo^T 2M
// ---------------------------------------------------------------------------
extern "C" void kernel_launch(void* const* d_in, const int* in_sizes, int n_in,
                              void* d_out, int out_size, void* d_ws,
                              size_t ws_size, hipStream_t stream) {
  (void)in_sizes; (void)n_in; (void)out_size; (void)ws_size;
  const float* x  = (const float*)d_in[0];
  const float* Wq = (const float*)d_in[1];
  const float* Wk = (const float*)d_in[2];
  const float* Wv = (const float*)d_in[3];
  const float* Wo = (const float*)d_in[4];
  const float* bo = (const float*)d_in[5];
  float* out = (float*)d_out;

  char* ws = (char*)d_ws;
  const size_t MB = 1024 * 1024;
  unsigned short* Qw  = (unsigned short*)(ws);
  unsigned short* Kw  = (unsigned short*)(ws + 8 * MB);
  unsigned short* Vt  = (unsigned short*)(ws + 16 * MB);
  unsigned short* ab  = (unsigned short*)(ws + 24 * MB);
  unsigned short* Wqt = (unsigned short*)(ws + 32 * MB);
  unsigned short* Wkt = Wqt + HH * HD * HD;
  unsigned short* Wvt = Wkt + HH * HD * HD;
  unsigned short* WoT = Wvt + HH * HD * HD;

  cvt_t64<<<256, 256, 0, stream>>>(Wq, Wqt);
  cvt_t64<<<256, 256, 0, stream>>>(Wk, Wkt);
  cvt_t64<<<256, 256, 0, stream>>>(Wv, Wvt);
  cvt_tD<<<4096, 256, 0, stream>>>(Wo, WoT);

  qkv_proj<<<512, 256, 0, stream>>>(x, Wqt, Wkt, Wvt, Qw, Kw, Vt);  // 4096 waves
  flash_attn<<<256, 256, 0, stream>>>(Qw, Kw, Vt, ab);              // 2048 waves
  out_proj<<<512, 256, 0, stream>>>(ab, WoT, bo, out);              // 4096 waves
}
